// DeformFuser_6468220748303
// MI455X (gfx1250) — matrix operand order repaired and checked by reading, not compiled, not run on hardware
//
#include <hip/hip_runtime.h>
#include <math.h>

// ---------------------------------------------------------------------------
// MI455X (gfx1250) fused implementation.
//   - f16 WMMA (v_wmma_f32_16x16x32_f16), f32 accumulate
//   - implicit-GEMM 3x3 convs (one K=32 WMMA step per tap)
//   - weights pre-packed into B-fragment order (contiguous k = 16*(lane/16) + s)
//   - async global->LDS staging (GLOBAL_LOAD_ASYNC_TO_LDS_B128 + ASYNCcnt)
//   - branchless sigmoid-gelu (erff lowers to exec-masked branchy polys)
// ---------------------------------------------------------------------------

typedef _Float16 h16;
typedef __attribute__((ext_vector_type(16))) _Float16 v16h;
typedef __attribute__((ext_vector_type(8)))  float    v8f;

#define NBATCH 2
#define NVIEW  4
#define NIMG   8
#define IPIX   (128*128)
#define OPIX   (256*256)

// packed B-tile indices (each tile = 512 halves = 32 lanes x 16 halves)
#define T_FD1   0     // 16 tiles (K=64 ->2, N=128 ->8)
#define T_FD2   16    // 16 tiles (K=128->4, N=64 ->4)
#define T_BC1   32    // 24 tiles (K=96 ->3, N=128->8)
#define T_BC2   56    // 8  tiles (K=128->4, N=32 ->2)
#define T_CD1   64    // 4  tiles (K=54->2,  N=32 ->2)
#define T_CD21  68    // 18 tiles (9 taps x 2 N)
#define T_CD22  86
#define T_CD24  104
#define T_CD3   122   // 2 tiles
#define T_GGV   124   // 9 tiles (9 taps, N=16: 0..8 guide, 9 gview)
#define T_GATE  133   // 9 tiles (N=16: 0 gate)
#define T_TOTAL 142

// bias offsets in float region
#define BO_FD1  0
#define BO_FD2  128
#define BO_BC1  192
#define BO_BC2  320
#define BO_CD1  352
#define BO_CD21 384
#define BO_CD22 416
#define BO_CD24 448
#define BO_CD3  480
#define BO_GGV  512
#define BO_GATE 528
#define BIAS_N  544

// workspace byte offsets
#define WS_PK   0u
#define WS_BIAS 145408u
#define WS_PC   147584u       // NIMG*3*OPIX f32  (6291456 B)
#define WS_PM   6439040u      // NIMG*OPIX   f32  (2097152 B)
#define WS_X    8536192u      // NIMG*OPIX*32 h16 (33554432 B)
#define WS_DC   42090624u     // NIMG*OPIX*32 h16
#define WS_GU   75645056u     // NIMG*OPIX*32 h16
// end ~ 109.2 MB

// ----------------------------- helpers -------------------------------------

__device__ __forceinline__ float geluf(float x){
  // branchless fast gelu: x * sigmoid(1.702 x)  (one v_exp_f32)
  return x / (1.0f + __expf(-1.702f * x));
}
__device__ __forceinline__ float sigm(float x){
  return 1.0f / (1.0f + __expf(-x));
}

__device__ __forceinline__ v8f wmma16(v16h a, v16h b, v8f c){
  return __builtin_amdgcn_wmma_f32_16x16x32_f16(false, a, false, b, (short)0, c, false, false);
}

// A fragment from an LDS half buffer. rowHalf = half-offset of this lane's
// pixel row (channel 0). K layout: VGPR p holds K = 2*p
// (+16 for lanes 16..31).
__device__ __forceinline__ v16h load_a_lds(const unsigned* U, int rowHalf, int kbase, int laneHi){
  union { v16h v; unsigned u[8]; } f;
#pragma unroll
  for (int p = 0; p < 8; ++p){
    int k = kbase + 2*p + 16*laneHi;
    f.u[p] = U[(rowHalf + k) >> 1];
  }
  return f.v;
}

// B fragment: pre-packed, lane-contiguous (16 halves per lane).
__device__ __forceinline__ v16h load_b_pk(const h16* pk, int tile, int lane){
  union { v16h v; uint4 q[2]; } f;
  const uint4* s = reinterpret_cast<const uint4*>(pk + ((size_t)tile << 9) + (lane << 4));
  f.q[0] = s[0]; f.q[1] = s[1];
  return f.v;
}

// Async DMA one 16B packet: LDS[lds+imm] <- MEM[base + goff + imm].
// (ISA 15.18: INST_OFFSET is added to both LDS and global addresses.)
__device__ __forceinline__ void async_ldsb128(unsigned lds, unsigned goff,
                                              const void* base, int imm){
  asm volatile("global_load_async_to_lds_b128 %0, %1, %2 offset:%c3"
               :: "v"(lds), "v"(goff), "s"(base), "n"(imm) : "memory");
}
__device__ __forceinline__ void wait_asynccnt0(){
  asm volatile("s_wait_asynccnt 0x0" ::: "memory");
}
// 32-bit LDS byte offset of a generic pointer to LDS (low 32 bits).
__device__ __forceinline__ unsigned lds_addr32(const void* p){
  return (unsigned)(size_t)p;
}

__device__ __forceinline__ void bilin(int o, int& i0, int& i1, float& w1){
  float f  = (o + 0.5f) * 0.5f - 0.5f;
  float fl = floorf(f);
  w1 = f - fl;
  int i = (int)fl;
  i0 = i < 0 ? 0 : (i > 127 ? 127 : i);
  int j = i + 1;
  i1 = j < 0 ? 0 : (j > 127 ? 127 : j);
}

// ----------------------------- weight packing ------------------------------

struct PackArgs {
  const float *fd1w,*fd1b,*fd2w,*fd2b,*bc1w,*bc1b,*bc2w,*bc2b;
  const float *cd1w,*cd1b,*cd21w,*cd21b,*cd22w,*cd22b,*cd24w,*cd24b,*cd3w,*cd3b;
  const float *gatew,*gateg,*gateb,*gatem,*gatev;
  const float *guidew,*guideg,*guideb,*guidem,*guidev;
  const float *gvieww,*gviewg,*gviewb,*gviewm,*gviewv;
};

__device__ void pack1x1(h16* dst, const float* w, int Kreal, int N, int Kt, int Nt){
  int total = Kt * Nt * 512;
  for (int i = threadIdx.x; i < total; i += blockDim.x){
    int tile = i >> 9, rem = i & 511, lane = rem >> 4, h = rem & 15;
    int nt = tile / Kt, ks = tile - nt * Kt;
    int k = ks * 32 + h + 16 * (lane >> 4);
    int n = nt * 16 + (lane & 15);
    float v = 0.f;
    if (k < Kreal && n < N) v = w[n * Kreal + k];
    dst[i] = (h16)v;
  }
}

__device__ void pack3x3(h16* dst, const float* w){ // 32-in, 32-out
  for (int i = threadIdx.x; i < 18 * 512; i += blockDim.x){
    int tile = i >> 9, rem = i & 511, lane = rem >> 4, h = rem & 15;
    int nt = tile / 9, tp = tile - nt * 9;
    int ci = h + 16 * (lane >> 4);
    int n  = nt * 16 + (lane & 15);
    dst[i] = (h16)w[(n * 32 + ci) * 9 + tp];
  }
}

__global__ __launch_bounds__(256) void k_pack(PackArgs pa, h16* pk, float* bias){
  pack1x1(pk + T_FD1 * 512, pa.fd1w,  64, 128, 2, 8);
  pack1x1(pk + T_FD2 * 512, pa.fd2w, 128,  64, 4, 4);
  pack1x1(pk + T_BC1 * 512, pa.bc1w,  96, 128, 3, 8);
  pack1x1(pk + T_BC2 * 512, pa.bc2w, 128,  32, 4, 2);
  pack1x1(pk + T_CD3 * 512, pa.cd3w,  32,  32, 1, 2);
  pack3x3(pk + T_CD21 * 512, pa.cd21w);
  pack3x3(pk + T_CD22 * 512, pa.cd22w);
  pack3x3(pk + T_CD24 * 512, pa.cd24w);
  // cd1: K index = tap*6 + ch, padded to 64
  for (int i = threadIdx.x; i < 4 * 512; i += blockDim.x){
    int tile = i >> 9, rem = i & 511, lane = rem >> 4, h = rem & 15;
    int nt = tile >> 1, ks = tile & 1;
    int k = ks * 32 + h + 16 * (lane >> 4);
    int n = nt * 16 + (lane & 15);
    float v = 0.f;
    if (k < 54){ int tp = k / 6, c = k - tp * 6; v = pa.cd1w[(n * 6 + c) * 9 + tp]; }
    pk[T_CD1 * 512 + i] = (h16)v;
  }
  // guide(9) + gview(1) packed together, BN scale folded
  for (int i = threadIdx.x; i < 9 * 512; i += blockDim.x){
    int tp = i >> 9, rem = i & 511, lane = rem >> 4, h = rem & 15;
    int ci = h + 16 * (lane >> 4);
    int n  = lane & 15;
    float v = 0.f;
    if (n < 9){
      float sc = pa.guideg[n] * rsqrtf(pa.guidev[n] + 1e-5f);
      v = pa.guidew[(n * 32 + ci) * 9 + tp] * sc;
    } else if (n == 9){
      float sc = pa.gviewg[0] * rsqrtf(pa.gviewv[0] + 1e-5f);
      v = pa.gvieww[ci * 9 + tp] * sc;
    }
    pk[T_GGV * 512 + i] = (h16)v;
  }
  // gate, BN folded, N=16 with channel 0 used
  for (int i = threadIdx.x; i < 9 * 512; i += blockDim.x){
    int tp = i >> 9, rem = i & 511, lane = rem >> 4, h = rem & 15;
    int ci = h + 16 * (lane >> 4);
    int n  = lane & 15;
    float v = 0.f;
    if (n == 0){
      float sc = pa.gateg[0] * rsqrtf(pa.gatev[0] + 1e-5f);
      v = pa.gatew[ci * 9 + tp] * sc;
    }
    pk[T_GATE * 512 + i] = (h16)v;
  }
  // biases
  for (int i = threadIdx.x; i < 128; i += blockDim.x) bias[BO_FD1 + i] = pa.fd1b[i];
  for (int i = threadIdx.x; i <  64; i += blockDim.x) bias[BO_FD2 + i] = pa.fd2b[i];
  for (int i = threadIdx.x; i < 128; i += blockDim.x) bias[BO_BC1 + i] = pa.bc1b[i];
  for (int i = threadIdx.x; i <  32; i += blockDim.x){
    bias[BO_BC2  + i] = pa.bc2b[i];
    bias[BO_CD1  + i] = pa.cd1b[i];
    bias[BO_CD21 + i] = pa.cd21b[i];
    bias[BO_CD22 + i] = pa.cd22b[i];
    bias[BO_CD24 + i] = pa.cd24b[i];
    bias[BO_CD3  + i] = pa.cd3b[i];
  }
  for (int i = threadIdx.x; i < 16; i += blockDim.x){
    float v = 0.f;
    if (i < 9){
      float sc = pa.guideg[i] * rsqrtf(pa.guidev[i] + 1e-5f);
      v = pa.guideb[i] - pa.guidem[i] * sc;
    } else if (i == 9){
      float sc = pa.gviewg[0] * rsqrtf(pa.gviewv[0] + 1e-5f);
      v = pa.gviewb[0] - pa.gviewm[0] * sc;
    }
    bias[BO_GGV + i] = v;
    float g = 0.f;
    if (i == 0){
      float sc = pa.gateg[0] * rsqrtf(pa.gatev[0] + 1e-5f);
      g = pa.gateb[0] - pa.gatem[0] * sc;
    }
    bias[BO_GATE + i] = g;
  }
}

// -------------------- K1: bilinear resize pc & pm --------------------------

__global__ __launch_bounds__(256) void k_resize(const float* __restrict__ pcol,
                                                const float* __restrict__ pmsk,
                                                float* __restrict__ pcF,
                                                float* __restrict__ pmF){
  int idx = blockIdx.x * 256 + threadIdx.x;       // NIMG*OPIX threads
  int b   = idx >> 16;
  int pix = idx & 65535;
  int oy = pix >> 8, ox = pix & 255;
  int y0, y1, x0, x1; float wy, wx;
  bilin(oy, y0, y1, wy);
  bilin(ox, x0, x1, wx);
  int n = b >> 2, v = b & 3;
  const float* src = pcol + ((size_t)v * NBATCH + n) * 3 * IPIX;
#pragma unroll
  for (int c = 0; c < 3; ++c){
    const float* s = src + (size_t)c * IPIX;
    float a = (1.f - wx) * s[y0 * 128 + x0] + wx * s[y0 * 128 + x1];
    float d = (1.f - wx) * s[y1 * 128 + x0] + wx * s[y1 * 128 + x1];
    pcF[((size_t)b * 3 + c) * OPIX + pix] = (1.f - wy) * a + wy * d;
  }
  const float* ms = pmsk + ((size_t)v * NBATCH + n) * IPIX;
  float a = (1.f - wx) * ms[y0 * 128 + x0] + wx * ms[y0 * 128 + x1];
  float d = (1.f - wx) * ms[y1 * 128 + x0] + wx * ms[y1 * 128 + x1];
  pmF[(size_t)b * OPIX + pix] = (1.f - wy) * a + wy * d;
}

// -------------------- K2: cd1 (3x3, 6->32) + gelu --------------------------

__global__ __launch_bounds__(256) void k_cd1(const float* __restrict__ deep,
                                             const float* __restrict__ bias,
                                             const h16* __restrict__ pk,
                                             const float* __restrict__ pcF,
                                             h16* __restrict__ xH){
  __shared__ __align__(16) h16 A0[256 * 64];
  int blk = blockIdx.x;
  int b = blk >> 8, t = blk & 255, ty = t >> 4, tx = t & 15;
  int tid = threadIdx.x;
  int n = b >> 2;
  {
    int px = tid, py = px >> 4, pxx = px & 15;
    int gy = ty * 16 + py, gx = tx * 16 + pxx;
#pragma unroll
    for (int tap = 0; tap < 9; ++tap){
      int sy = gy + tap / 3 - 1, sx = gx + tap % 3 - 1;
      bool inb = ((unsigned)sy < 256u) && ((unsigned)sx < 256u);
      int sp = sy * 256 + sx;
#pragma unroll
      for (int c = 0; c < 6; ++c){
        float v = 0.f;
        if (inb) v = (c < 3) ? deep[((size_t)n * 3 + c) * OPIX + sp]
                             : pcF[((size_t)b * 3 + (c - 3)) * OPIX + sp];
        A0[px * 64 + tap * 6 + c] = (h16)v;
      }
    }
#pragma unroll
    for (int k = 54; k < 64; ++k) A0[px * 64 + k] = (h16)0.f;
  }
  __syncthreads();
  int wv = tid >> 5, lane = tid & 31, laneHi = lane >> 4, lm = lane & 15;
  const unsigned* U = (const unsigned*)A0;
  v16h b00 = load_b_pk(pk, T_CD1 + 0, lane), b01 = load_b_pk(pk, T_CD1 + 1, lane);
  v16h b10 = load_b_pk(pk, T_CD1 + 2, lane), b11 = load_b_pk(pk, T_CD1 + 3, lane);
#pragma unroll
  for (int mt = 0; mt < 2; ++mt){
    int m = wv + 8 * mt;
    int rowOff = (m * 16 + lm) * 64;
    v16h a0 = load_a_lds(U, rowOff, 0,  laneHi);
    v16h a1 = load_a_lds(U, rowOff, 32, laneHi);
#pragma unroll
    for (int nt = 0; nt < 2; ++nt){
      v8f c = {};
      c = wmma16(a0, nt ? b10 : b00, c);
      c = wmma16(a1, nt ? b11 : b01, c);
      int ch = nt * 16 + lm;
      float bb = bias[BO_CD1 + ch];
#pragma unroll
      for (int r = 0; r < 8; ++r){
        int M = r + 8 * laneHi;
        int gpix = (ty * 16 + m) * 256 + tx * 16 + M;
        xH[((size_t)b * OPIX + gpix) * 32 + ch] = (h16)geluf(c[r] + bb);
      }
    }
  }
}

// -------- K3: cd2x{1,2,4} (3x3 dil 1/2/4) + gelu + sum + cd3 + gelu --------

__global__ __launch_bounds__(256) void k_cd2(const float* __restrict__ bias,
                                             const h16* __restrict__ pk,
                                             const h16* __restrict__ xH,
                                             h16* __restrict__ dcH){
  __shared__ __align__(16) h16 xt[576 * 32];   // 24x24 halo-4 tile
  __shared__ __align__(16) h16 sm[256 * 32];
  int blk = blockIdx.x;
  int b = blk >> 8, t = blk & 255, ty = t >> 4, tx = t & 15;
  int tid = threadIdx.x;
  const void* xbase = (const void*)(xH + (size_t)b * OPIX * 32);
  for (int hp = tid; hp < 576; hp += 256){
    int hy = hp / 24, hx = hp - hy * 24;
    int gy = ty * 16 + hy - 4, gx = tx * 16 + hx - 4;
    if ((unsigned)gy < 256u && (unsigned)gx < 256u){
      unsigned lds  = lds_addr32(xt + hp * 32);
      unsigned goff = (unsigned)(gy * 256 + gx) * 64u;
      async_ldsb128(lds, goff, xbase, 0);
      async_ldsb128(lds, goff, xbase, 16);
      async_ldsb128(lds, goff, xbase, 32);
      async_ldsb128(lds, goff, xbase, 48);
    } else {
      uint4 z; z.x = z.y = z.z = z.w = 0u;
      uint4* d = (uint4*)(xt + hp * 32);
      d[0] = z; d[1] = z; d[2] = z; d[3] = z;
    }
  }
  wait_asynccnt0();
  __syncthreads();
  int wv = tid >> 5, lane = tid & 31, laneHi = lane >> 4, lm = lane & 15;
  const unsigned* U = (const unsigned*)xt;
#pragma unroll
  for (int mt = 0; mt < 2; ++mt){
    int m = wv + 8 * mt;
    int py = m + 4, lx = lm + 4;
    v8f a10 = {}, a11 = {}, a20 = {}, a21 = {}, a40 = {}, a41 = {};
    auto conv3 = [&](int tbase, int dil, v8f& c0, v8f& c1){
#pragma unroll
      for (int tap = 0; tap < 9; ++tap){
        int dy = (tap / 3 - 1) * dil, dx = (tap % 3 - 1) * dil;
        int rowOff = ((py + dy) * 24 + (lx + dx)) * 32;
        v16h a = load_a_lds(U, rowOff, 0, laneHi);
        c0 = wmma16(a, load_b_pk(pk, tbase + tap,     lane), c0);
        c1 = wmma16(a, load_b_pk(pk, tbase + 9 + tap, lane), c1);
      }
    };
    conv3(T_CD21, 1, a10, a11);
    conv3(T_CD22, 2, a20, a21);
    conv3(T_CD24, 4, a40, a41);
#pragma unroll
    for (int nt = 0; nt < 2; ++nt){
      int ch = nt * 16 + lm;
      float b1 = bias[BO_CD21 + ch], b2 = bias[BO_CD22 + ch], b4 = bias[BO_CD24 + ch];
      v8f c1v = nt ? a11 : a10, c2v = nt ? a21 : a20, c4v = nt ? a41 : a40;
#pragma unroll
      for (int r = 0; r < 8; ++r){
        float s = geluf(c1v[r] + b1) + geluf(c2v[r] + b2) + geluf(c4v[r] + b4);
        int pixl = m * 16 + r + 8 * laneHi;
        sm[pixl * 32 + ch] = (h16)s;
      }
    }
  }
  __syncthreads();
  const unsigned* S = (const unsigned*)sm;
#pragma unroll
  for (int mt = 0; mt < 2; ++mt){
    int m = wv + 8 * mt;
    int rowOff = (m * 16 + lm) * 32;
    v16h a = load_a_lds(S, rowOff, 0, laneHi);
#pragma unroll
    for (int nt = 0; nt < 2; ++nt){
      v8f c = {};
      c = wmma16(a, load_b_pk(pk, T_CD3 + nt, lane), c);
      int ch = nt * 16 + lm;
      float bb = bias[BO_CD3 + ch];
#pragma unroll
      for (int r = 0; r < 8; ++r){
        int gpix = (ty * 16 + m) * 256 + tx * 16 + (r + 8 * laneHi);
        dcH[((size_t)b * OPIX + gpix) * 32 + ch] = (h16)geluf(c[r] + bb);
      }
    }
  }
}

// ---- K4: df resize -> fd1 -> fd2, concat dc, bc1 -> bc2 -> guidance -------
// tile = 128 pixels (8 rows x 16 cols), 8 waves, 1 M-tile each

__global__ __launch_bounds__(256) void k_mlp(const float* __restrict__ dfeats,
                                             const float* __restrict__ bias,
                                             const h16* __restrict__ pk,
                                             const h16* __restrict__ dcH,
                                             h16* __restrict__ guH){
  __shared__ __align__(16) h16 A0[128 * 64];   // resized df, later df(64)
  __shared__ __align__(16) h16 H1[128 * 128];  // hidden 128
  __shared__ __align__(16) h16 DC[128 * 32];   // dc channels
  int blk = blockIdx.x;
  int b = blk >> 9, t = blk & 511, ty = t >> 4, tx = t & 15;
  int tid = threadIdx.x;
  __builtin_prefetch(pk + T_FD1 * 512, 0, 0);
  // async DMA of the dc tile straight into LDS (no OOB possible)
  {
    const void* dbase = (const void*)(dcH + (size_t)b * OPIX * 32);
    for (int i = tid; i < 128; i += 256){    // one 64B row per lane
      int gpix = (ty * 8 + (i >> 4)) * 256 + tx * 16 + (i & 15);
      unsigned lds  = lds_addr32(DC + i * 32);
      unsigned goff = (unsigned)gpix * 64u;
      async_ldsb128(lds, goff, dbase, 0);
      async_ldsb128(lds, goff, dbase, 16);
      async_ldsb128(lds, goff, dbase, 32);
      async_ldsb128(lds, goff, dbase, 48);
    }
  }
  {
    int px = tid & 127, half = tid >> 7;
    int py = px >> 4, pxx = px & 15;
    int gy = ty * 8 + py, gx = tx * 16 + pxx;
    int y0, y1, x0, x1; float wy, wx;
    bilin(gy, y0, y1, wy);
    bilin(gx, x0, x1, wx);
    const float* src = dfeats + (size_t)b * 64 * IPIX;
    int c0 = half * 32;
#pragma unroll 4
    for (int c = 0; c < 32; ++c){
      const float* s = src + (size_t)(c0 + c) * IPIX;
      float a = (1.f - wx) * s[y0 * 128 + x0] + wx * s[y0 * 128 + x1];
      float d = (1.f - wx) * s[y1 * 128 + x0] + wx * s[y1 * 128 + x1];
      A0[px * 64 + c0 + c] = (h16)((1.f - wy) * a + wy * d);
    }
  }
  wait_asynccnt0();
  __syncthreads();
  int wv = tid >> 5, lane = tid & 31, laneHi = lane >> 4, lm = lane & 15;
  int m = wv;
  const unsigned* UA = (const unsigned*)A0;
  const unsigned* UH = (const unsigned*)H1;
  const unsigned* UD = (const unsigned*)DC;
  // fd1: 64 -> 128
  {
    int rowOff = (m * 16 + lm) * 64;
    v16h a0 = load_a_lds(UA, rowOff, 0,  laneHi);
    v16h a1 = load_a_lds(UA, rowOff, 32, laneHi);
#pragma unroll
    for (int nt = 0; nt < 8; ++nt){
      v8f c = {};
      c = wmma16(a0, load_b_pk(pk, T_FD1 + nt * 2 + 0, lane), c);
      c = wmma16(a1, load_b_pk(pk, T_FD1 + nt * 2 + 1, lane), c);
      float bb = bias[BO_FD1 + nt * 16 + lm];
#pragma unroll
      for (int r = 0; r < 8; ++r)
        H1[(m * 16 + r + 8 * laneHi) * 128 + nt * 16 + lm] = (h16)geluf(c[r] + bb);
    }
  }
  __syncthreads();
  // fd2: 128 -> 64 (into A0)
  {
    int rowOff = (m * 16 + lm) * 128;
    v16h a[4];
#pragma unroll
    for (int ks = 0; ks < 4; ++ks) a[ks] = load_a_lds(UH, rowOff, ks * 32, laneHi);
#pragma unroll
    for (int nt = 0; nt < 4; ++nt){
      v8f c = {};
#pragma unroll
      for (int ks = 0; ks < 4; ++ks)
        c = wmma16(a[ks], load_b_pk(pk, T_FD2 + nt * 4 + ks, lane), c);
      float bb = bias[BO_FD2 + nt * 16 + lm];
#pragma unroll
      for (int r = 0; r < 8; ++r)
        A0[(m * 16 + r + 8 * laneHi) * 64 + nt * 16 + lm] = (h16)geluf(c[r] + bb);
    }
  }
  __syncthreads();
  // bc1: concat(df=A0[0..63], dc=DC[0..31]) -> 128
  {
    int rowOff = (m * 16 + lm) * 64;
    int rowOffD = (m * 16 + lm) * 32;
    v16h a0 = load_a_lds(UA, rowOff, 0,  laneHi);
    v16h a1 = load_a_lds(UA, rowOff, 32, laneHi);
    v16h a2 = load_a_lds(UD, rowOffD, 0, laneHi);
#pragma unroll
    for (int nt = 0; nt < 8; ++nt){
      v8f c = {};
      c = wmma16(a0, load_b_pk(pk, T_BC1 + nt * 3 + 0, lane), c);
      c = wmma16(a1, load_b_pk(pk, T_BC1 + nt * 3 + 1, lane), c);
      c = wmma16(a2, load_b_pk(pk, T_BC1 + nt * 3 + 2, lane), c);
      float bb = bias[BO_BC1 + nt * 16 + lm];
#pragma unroll
      for (int r = 0; r < 8; ++r)
        H1[(m * 16 + r + 8 * laneHi) * 128 + nt * 16 + lm] = (h16)geluf(c[r] + bb);
    }
  }
  __syncthreads();
  // bc2: 128 -> 32 (guidance out)
  {
    int rowOff = (m * 16 + lm) * 128;
    v16h a[4];
#pragma unroll
    for (int ks = 0; ks < 4; ++ks) a[ks] = load_a_lds(UH, rowOff, ks * 32, laneHi);
#pragma unroll
    for (int nt = 0; nt < 2; ++nt){
      v8f c = {};
#pragma unroll
      for (int ks = 0; ks < 4; ++ks)
        c = wmma16(a[ks], load_b_pk(pk, T_BC2 + nt * 4 + ks, lane), c);
      int ch = nt * 16 + lm;
      float bb = bias[BO_BC2 + ch];
#pragma unroll
      for (int r = 0; r < 8; ++r){
        int gpix = (ty * 8 + m) * 256 + tx * 16 + (r + 8 * laneHi);
        guH[((size_t)b * OPIX + gpix) * 32 + ch] = (h16)geluf(c[r] + bb);
      }
    }
  }
}

// -------- K5: heads (gate/gview/guide), guided filter, fusion, outputs -----

__global__ __launch_bounds__(256) void k_final(const float* __restrict__ deep,
                                               const float* __restrict__ bias,
                                               const h16* __restrict__ pk,
                                               const h16* __restrict__ guH,
                                               const float* __restrict__ pcF,
                                               const float* __restrict__ pmF,
                                               float* __restrict__ out){
  __shared__ __align__(16) h16  gA[324 * 32];   // 18x18 halo-1 guidance tile
  __shared__ __align__(16) h16  gM[324 * 32];   // running mean over views
  __shared__ __align__(16) float cv[256 * 10];  // guide(9) + gview conv out
  __shared__ __align__(16) float gt[256];       // gate conv out
  int blk = blockIdx.x;
  int n = blk >> 8, t = blk & 255, ty = t >> 4, tx = t & 15;
  int tid = threadIdx.x;
  int wv = tid >> 5, lane = tid & 31, laneHi = lane >> 4, lm = lane & 15;
  int px = tid, py = px >> 4, pxx = px & 15;
  int gy = ty * 16 + py, gx = tx * 16 + pxx;
  int gpix = gy * 256 + gx;

  for (int i = tid; i < 324 * 32; i += 256) gM[i] = (h16)0.f;
  float modsum0 = 0.f, modsum1 = 0.f, modsum2 = 0.f, gvsum = 0.f, pmsum = 0.f;
  __syncthreads();

  for (int v = 0; v < 4; ++v){
    int b = n * 4 + v;
    const void* gbase = (const void*)(guH + (size_t)b * OPIX * 32);
    for (int hp = tid; hp < 324; hp += 256){
      int hy = hp / 18, hx = hp - hy * 18;
      int sy = ty * 16 + hy - 1, sx = tx * 16 + hx - 1;
      if ((unsigned)sy < 256u && (unsigned)sx < 256u){
        unsigned lds  = lds_addr32(gA + hp * 32);
        unsigned goff = (unsigned)(sy * 256 + sx) * 64u;
        async_ldsb128(lds, goff, gbase, 0);
        async_ldsb128(lds, goff, gbase, 16);
        async_ldsb128(lds, goff, gbase, 32);
        async_ldsb128(lds, goff, gbase, 48);
      } else {
        uint4 z; z.x = z.y = z.z = z.w = 0u;
        uint4* d = (uint4*)(gA + hp * 32);
        d[0] = z; d[1] = z; d[2] = z; d[3] = z;
      }
    }
    wait_asynccnt0();
    __syncthreads();
    for (int i = tid; i < 324 * 32; i += 256)
      gM[i] = (h16)((float)gM[i] + 0.25f * (float)gA[i]);
    // guide+gview conv (WMMA, N=16: ch 0..8 guide, 9 gview)
    const unsigned* U = (const unsigned*)gA;
#pragma unroll
    for (int mt = 0; mt < 2; ++mt){
      int m = wv + 8 * mt;
      v8f c = {};
#pragma unroll
      for (int tap = 0; tap < 9; ++tap){
        int rowOff = ((m + tap / 3) * 18 + (lm + tap % 3)) * 32;
        c = wmma16(load_a_lds(U, rowOff, 0, laneHi), load_b_pk(pk, T_GGV + tap, lane), c);
      }
      float bb = bias[BO_GGV + lm];
      if (lm < 10){
#pragma unroll
        for (int r = 0; r < 8; ++r)
          cv[(m * 16 + r + 8 * laneHi) * 10 + lm] = c[r] + bb;
      }
    }
    __syncthreads();
    // scalar per-pixel: softmax, guided filter, accumulate
    {
      float g[9];
      float mx = -1e30f;
#pragma unroll
      for (int k = 0; k < 9; ++k){ g[k] = cv[px * 10 + k]; mx = fmaxf(mx, g[k]); }
      float ssum = 0.f;
#pragma unroll
      for (int k = 0; k < 9; ++k){ g[k] = __expf(g[k] - mx); ssum += g[k]; }
      float inv = 1.f / ssum;
#pragma unroll
      for (int k = 0; k < 9; ++k) g[k] *= inv;
      float pmv = pmF[(size_t)b * OPIX + gpix];
      float gv  = sigm(cv[px * 10 + 9]) * pmv;
      float gmid = g[4];
      float mods[3];
#pragma unroll
      for (int c3 = 0; c3 < 3; ++c3){
        const float* pcc = pcF + ((size_t)b * 3 + c3) * OPIX;
        float full = 0.f;
#pragma unroll
        for (int k = 0; k < 9; ++k){
          int sy = gy + k / 3 - 1, sx = gx + k % 3 - 1;
          float pv = ((unsigned)sy < 256u && (unsigned)sx < 256u) ? pcc[sy * 256 + sx] : 0.f;
          full += g[k] * pv;
        }
        float ctr = pcc[gpix];
        float base = full - gmid * ctr;
        float mm = ctr;
#pragma unroll
        for (int it = 0; it < 6; ++it) mm = base + gmid * mm;
        mods[c3] = mm * gv;
      }
      modsum0 += mods[0]; modsum1 += mods[1]; modsum2 += mods[2];
      gvsum += gv; pmsum += pmv;
    }
    __syncthreads();
  }
  // gate conv on mean guidance
  {
    const unsigned* U = (const unsigned*)gM;
#pragma unroll
    for (int mt = 0; mt < 2; ++mt){
      int m = wv + 8 * mt;
      v8f c = {};
#pragma unroll
      for (int tap = 0; tap < 9; ++tap){
        int rowOff = ((m + tap / 3) * 18 + (lm + tap % 3)) * 32;
        c = wmma16(load_a_lds(U, rowOff, 0, laneHi), load_b_pk(pk, T_GATE + tap, lane), c);
      }
      if (lm == 0){
#pragma unroll
        for (int r = 0; r < 8; ++r)
          gt[m * 16 + r + 8 * laneHi] = c[r] + bias[BO_GATE];
      }
    }
  }
  __syncthreads();
  {
    float gating = sigm(gt[px]) * (pmsum > 0.f ? 1.f : 0.f);
    float invg = 1.f / (gvsum + 1e-7f);
    float ms[3] = { modsum0 * invg, modsum1 * invg, modsum2 * invg };
#pragma unroll
    for (int c3 = 0; c3 < 3; ++c3){
      float dcv = deep[((size_t)n * 3 + c3) * OPIX + gpix];
      out[((size_t)n * 3 + c3) * OPIX + gpix] =
          (ms[c3] * gating + dcv * (1.f - gating)) * 2.f - 1.f;
      out[(size_t)NBATCH * 3 * OPIX + ((size_t)n * 3 + c3) * OPIX + gpix] =
          ms[c3] * 2.f - 1.f;
    }
  }
}

// ----------------------------- launcher ------------------------------------

extern "C" void kernel_launch(void* const* d_in, const int* in_sizes, int n_in,
                              void* d_out, int out_size, void* d_ws, size_t ws_size,
                              hipStream_t stream) {
  const float* diff_feats = (const float*)d_in[0];
  const float* deep_color = (const float*)d_in[1];
  const float* proj_col   = (const float*)d_in[2];
  const float* proj_msk   = (const float*)d_in[3];

  // params flattened in jax pytree order (dict keys sorted):
  // bc1, bc2, cd1, cd2x1, cd2x2, cd2x4, cd3, fd1, fd2, gate(w,g,b,m,v),
  // guide(w,g,b,m,v), gview(w,g,b,m,v)
  PackArgs pa;
  pa.bc1w  = (const float*)d_in[4];  pa.bc1b  = (const float*)d_in[5];
  pa.bc2w  = (const float*)d_in[6];  pa.bc2b  = (const float*)d_in[7];
  pa.cd1w  = (const float*)d_in[8];  pa.cd1b  = (const float*)d_in[9];
  pa.cd21w = (const float*)d_in[10]; pa.cd21b = (const float*)d_in[11];
  pa.cd22w = (const float*)d_in[12]; pa.cd22b = (const float*)d_in[13];
  pa.cd24w = (const float*)d_in[14]; pa.cd24b = (const float*)d_in[15];
  pa.cd3w  = (const float*)d_in[16]; pa.cd3b  = (const float*)d_in[17];
  pa.fd1w  = (const float*)d_in[18]; pa.fd1b  = (const float*)d_in[19];
  pa.fd2w  = (const float*)d_in[20]; pa.fd2b  = (const float*)d_in[21];
  pa.gatew = (const float*)d_in[22]; pa.gateg = (const float*)d_in[23];
  pa.gateb = (const float*)d_in[24]; pa.gatem = (const float*)d_in[25];
  pa.gatev = (const float*)d_in[26];
  pa.guidew = (const float*)d_in[27]; pa.guideg = (const float*)d_in[28];
  pa.guideb = (const float*)d_in[29]; pa.guidem = (const float*)d_in[30];
  pa.guidev = (const float*)d_in[31];
  pa.gvieww = (const float*)d_in[32]; pa.gviewg = (const float*)d_in[33];
  pa.gviewb = (const float*)d_in[34]; pa.gviewm = (const float*)d_in[35];
  pa.gviewv = (const float*)d_in[36];

  char* ws = (char*)d_ws;
  h16*   pk   = (h16*)  (ws + WS_PK);
  float* bias = (float*)(ws + WS_BIAS);
  float* pcF  = (float*)(ws + WS_PC);
  float* pmF  = (float*)(ws + WS_PM);
  h16*   xH   = (h16*)  (ws + WS_X);
  h16*   dcH  = (h16*)  (ws + WS_DC);
  h16*   guH  = (h16*)  (ws + WS_GU);
  float* out  = (float*)d_out;

  hipLaunchKernelGGL(k_pack,   dim3(1),    dim3(256), 0, stream, pa, pk, bias);
  hipLaunchKernelGGL(k_resize, dim3(2048), dim3(256), 0, stream, proj_col, proj_msk, pcF, pmF);
  hipLaunchKernelGGL(k_cd1,    dim3(2048), dim3(256), 0, stream, deep_color, bias, pk, pcF, xH);
  hipLaunchKernelGGL(k_cd2,    dim3(2048), dim3(256), 0, stream, bias, pk, xH, dcH);
  hipLaunchKernelGGL(k_mlp,    dim3(4096), dim3(256), 0, stream, diff_feats, bias, pk, dcH, guH);
  hipLaunchKernelGGL(k_final,  dim3(512),  dim3(256), 0, stream, deep_color, bias, pk, guH, pcF, pmF, out);
}
